// TinyMoE_2027224563963
// MI455X (gfx1250) — compile-verified, matrix operand order checked by reading, not run on hardware
//
#include <hip/hip_runtime.h>

// ---------------------------------------------------------------------------
// TinyMoE on MI455X (gfx1250): grouped top-2 expert GEMM in bf16 WMMA.
//  - router + top-2 in fp32 (exact, memory-bound)
//  - expert matmuls grouped by expert, bf16 inputs / f32 accumulate via
//    v_wmma_f32_16x16x32_bf16; per-wave tile 16x128 (8 acc chains) for ILP
//  - combine gathers each token's 2 permuted rows in fixed order -> bitwise
//    deterministic output regardless of atomic slot ordering.
// ---------------------------------------------------------------------------

typedef __attribute__((ext_vector_type(16))) __bf16 bf16x16;
typedef __attribute__((ext_vector_type(8)))  __bf16 bf16x8;
typedef __attribute__((ext_vector_type(8)))  float  f32x8;
typedef __attribute__((ext_vector_type(4)))  float  f32x4;   // clang vector: OK for nontemporal builtins

constexpr int NTOK     = 4 * 4096;          // B*S tokens
constexpr int H        = 512;
constexpr int E        = 8;
constexpr int PAD_ROWS = 2 * NTOK + E * 16; // per-expert segments padded to 16
constexpr int ROW_TILES = PAD_ROWS / 16;    // 2056

__device__ __forceinline__ unsigned short f2bf(float x) {
  unsigned u = __float_as_uint(x);
  unsigned r = u + 0x7FFFu + ((u >> 16) & 1u);   // round-to-nearest-even
  return (unsigned short)(r >> 16);
}

// --------------------------- init: counts=0, row2token=-1 -------------------
__global__ __launch_bounds__(256) void moe_init(int* __restrict__ row2token,
                                                int* __restrict__ counts) {
  int i = blockIdx.x * 256 + threadIdx.x;
  if (i < PAD_ROWS) row2token[i] = -1;
  if (i < E) counts[i] = 0;
}

// --------------------------- router (wave-per-token) ------------------------
__global__ __launch_bounds__(256) void moe_router(
    const float* __restrict__ x, const float* __restrict__ rw,
    unsigned short* __restrict__ xb, int* __restrict__ topk_idx,
    float* __restrict__ topk_w, int* __restrict__ counts) {
  __shared__ float s_rw[E * H];
  int tid = threadIdx.x;
#pragma unroll
  for (int i = 0; i < 16; ++i) s_rw[tid * 16 + i] = rw[tid * 16 + i];
  __syncthreads();

  int wave = tid >> 5, lane = tid & 31;
  int t = blockIdx.x * 8 + wave;

  const float* xr = x + (size_t)t * H + lane * 16;
  float v[16];
#pragma unroll
  for (int j = 0; j < 4; ++j) {
    f32x4 a = reinterpret_cast<const f32x4*>(xr)[j];
    v[4 * j + 0] = a.x; v[4 * j + 1] = a.y;
    v[4 * j + 2] = a.z; v[4 * j + 3] = a.w;
  }

  // fused fp32 -> bf16 conversion of x (one pass over x total)
  unsigned pk[8];
#pragma unroll
  for (int j = 0; j < 8; ++j)
    pk[j] = (unsigned)f2bf(v[2 * j]) | ((unsigned)f2bf(v[2 * j + 1]) << 16);
  uint4* dst = reinterpret_cast<uint4*>(xb + (size_t)t * H + lane * 16);
  dst[0] = make_uint4(pk[0], pk[1], pk[2], pk[3]);
  dst[1] = make_uint4(pk[4], pk[5], pk[6], pk[7]);

  float part[E];
#pragma unroll
  for (int e = 0; e < E; ++e) {
    const float* wr = &s_rw[e * H + lane * 16];
    float s = 0.f;
#pragma unroll
    for (int j = 0; j < 16; ++j) s += v[j] * wr[j];
    part[e] = s;
  }
#pragma unroll
  for (int e = 0; e < E; ++e) {
    float s = part[e];
#pragma unroll
    for (int off = 16; off > 0; off >>= 1) s += __shfl_xor(s, off, 32);
    part[e] = s;
  }

  if (lane == 0) {
    float mx = part[0];
#pragma unroll
    for (int e = 1; e < E; ++e) mx = fmaxf(mx, part[e]);
    float ex[E], sum = 0.f;
#pragma unroll
    for (int e = 0; e < E; ++e) { ex[e] = __expf(part[e] - mx); sum += ex[e]; }
    int i0 = 0; float b0 = ex[0];
#pragma unroll
    for (int e = 1; e < E; ++e) if (ex[e] > b0) { b0 = ex[e]; i0 = e; }
    int i1 = (i0 == 0) ? 1 : 0; float b1 = ex[i1];
#pragma unroll
    for (int e = 0; e < E; ++e)
      if (e != i0 && ex[e] > b1) { b1 = ex[e]; i1 = e; }
    float p0 = b0 / sum, p1 = b1 / sum;
    float inv = 1.f / (p0 + p1 + 1e-6f);
    topk_idx[2 * t] = i0;  topk_idx[2 * t + 1] = i1;
    topk_w[2 * t]  = p0 * inv; topk_w[2 * t + 1] = p1 * inv;
    atomicAdd(&counts[i0], 1);
    atomicAdd(&counts[i1], 1);
  }
}

// --------------- expert weights: transpose + convert to bf16 ----------------
// Wt[e][o][h] = bf16(W[e][h][o]) so WMMA B-fragments load contiguously.
__global__ __launch_bounds__(256) void moe_wconv(const float* __restrict__ w,
                                                 unsigned short* __restrict__ wt) {
  __shared__ float tile[32][33];
  int e = blockIdx.z;
  int tx = threadIdx.x, ty = threadIdx.y;      // (32, 8)
  const float* we = w + (size_t)e * H * H;
  unsigned short* wte = wt + (size_t)e * H * H;
  int o0 = blockIdx.x * 32, h0 = blockIdx.y * 32;
#pragma unroll
  for (int i = 0; i < 4; ++i)
    tile[ty + i * 8][tx] = we[(size_t)(h0 + ty + i * 8) * H + o0 + tx];
  __syncthreads();
#pragma unroll
  for (int i = 0; i < 4; ++i)
    wte[(size_t)(o0 + ty + i * 8) * H + h0 + tx] = f2bf(tile[tx][ty + i * 8]);
}

// --------------------------- scan (tiny) ------------------------------------
__global__ void moe_scan(const int* __restrict__ counts, int* __restrict__ off,
                         int* __restrict__ cursor) {
  if (threadIdx.x == 0 && blockIdx.x == 0) {
    int acc = 0;
    for (int e = 0; e < E; ++e) {
      off[e] = acc; cursor[e] = acc;
      acc += (counts[e] + 15) & ~15;            // 16-align each expert segment
    }
    off[E] = acc;
  }
}

// --------------------------- scatter ----------------------------------------
__global__ __launch_bounds__(256) void moe_scatter(
    const int* __restrict__ topk_idx, int* __restrict__ cursor,
    int* __restrict__ row2token, int* __restrict__ tokrow) {
  int t = blockIdx.x * 256 + threadIdx.x;
  if (t >= NTOK) return;
#pragma unroll
  for (int k = 0; k < 2; ++k) {
    int e = topk_idx[2 * t + k];
    int slot = atomicAdd(&cursor[e], 1);
    row2token[slot] = t;
    tokrow[2 * t + k] = slot;
  }
}

// --------------------------- grouped expert GEMM ----------------------------
// 4 waves/block; each wave computes 16 (permuted token rows) x 128 (out cols)
// in f32: 8 accumulator chains -> 8 independent WMMAs per K-step, A-fragment
// amortized 8x. Block covers all 512 output columns.
__global__ __launch_bounds__(128) void moe_gemm(
    const unsigned short* __restrict__ xb, const unsigned short* __restrict__ wt,
    const int* __restrict__ off, const int* __restrict__ row2token,
    float* __restrict__ yp) {
  int lane = threadIdx.x & 31;
  int wv = threadIdx.x >> 5;
  int r0 = blockIdx.x * 16;
  if (r0 >= off[E]) return;
  int e = 0;
#pragma unroll
  for (int i = 0; i < E; ++i) if (r0 >= off[i + 1]) e = i + 1;

  int n0 = wv * 128;
  int m  = lane & 15;
  int kb = (lane >> 4) * 8;   // ISA 16-bit A/B fragment striping: hi half-wave
                              // holds K+8..15 / K+24..31 of each 32-chunk
  int tok = row2token[r0 + m];
  if (tok < 0) tok = 0;       // padding row: compute garbage, never read back
  const unsigned short* arow = xb + (size_t)tok * H;
  const unsigned short* brow[8];
#pragma unroll
  for (int j = 0; j < 8; ++j)
    brow[j] = wt + ((size_t)e * H + (n0 + j * 16 + m)) * H;

  f32x8 acc[8];
#pragma unroll
  for (int j = 0; j < 8; ++j)
    acc[j] = (f32x8){0.f, 0.f, 0.f, 0.f, 0.f, 0.f, 0.f, 0.f};

  for (int k0 = 0; k0 < H; k0 += 32) {
    union frag { bf16x16 v; bf16x8 h[2]; };
    frag afr;
    afr.h[0] = *reinterpret_cast<const bf16x8*>(arow + k0 + kb);
    afr.h[1] = *reinterpret_cast<const bf16x8*>(arow + k0 + kb + 16);
    frag bfr[8];
#pragma unroll
    for (int j = 0; j < 8; ++j) {
      bfr[j].h[0] = *reinterpret_cast<const bf16x8*>(brow[j] + k0 + kb);
      bfr[j].h[1] = *reinterpret_cast<const bf16x8*>(brow[j] + k0 + kb + 16);
    }
#pragma unroll
    for (int j = 0; j < 8; ++j)
      acc[j] = __builtin_amdgcn_wmma_f32_16x16x32_bf16(
          false, afr.v, false, bfr[j].v, (short)0, acc[j], false, false);
  }

  int rbase = r0 + ((lane >> 4) << 3);   // C/D layout: hi half-wave = M+8
#pragma unroll
  for (int j = 0; j < 8; ++j) {
    int col = n0 + j * 16 + m;
#pragma unroll
    for (int v = 0; v < 8; ++v)
      yp[(size_t)(rbase + v) * H + col] = acc[j][v];
  }
}

// --------------------------- combine ----------------------------------------
__global__ __launch_bounds__(128) void moe_combine(
    const float* __restrict__ yp, const int* __restrict__ tokrow,
    const float* __restrict__ tw, float* __restrict__ out) {
  int t = blockIdx.x;
  int c = threadIdx.x * 4;
  int r0 = tokrow[2 * t], r1 = tokrow[2 * t + 1];
  float w0 = tw[2 * t], w1 = tw[2 * t + 1];
  f32x4 a = *reinterpret_cast<const f32x4*>(yp + (size_t)r0 * H + c);
  f32x4 b = *reinterpret_cast<const f32x4*>(yp + (size_t)r1 * H + c);
  f32x4 o = w0 * a + w1 * b;
  // out is never re-read on device: bypass cache retention (non-temporal)
  __builtin_nontemporal_store(o, reinterpret_cast<f32x4*>(out + (size_t)t * H + c));
}

// ---------------------------------------------------------------------------
extern "C" void kernel_launch(void* const* d_in, const int* in_sizes, int n_in,
                              void* d_out, int out_size, void* d_ws,
                              size_t ws_size, hipStream_t stream) {
  const float* x  = (const float*)d_in[0];   // [N, H] fp32
  const float* rw = (const float*)d_in[1];   // [E, H] fp32
  const float* w  = (const float*)d_in[2];   // [E, H, H] fp32
  float* out = (float*)d_out;                // [N, H] fp32
  (void)in_sizes; (void)n_in; (void)out_size; (void)ws_size;

  char* ws = (char*)d_ws;
  size_t o = 0;
  auto alloc = [&](size_t bytes) -> char* {
    char* p = ws + o;
    o = (o + bytes + 255) & ~(size_t)255;
    return p;
  };
  unsigned short* xb        = (unsigned short*)alloc((size_t)NTOK * H * 2);
  unsigned short* wt        = (unsigned short*)alloc((size_t)E * H * H * 2);
  float*          yp        = (float*)alloc((size_t)PAD_ROWS * H * 4);
  int*            topk_idx  = (int*)alloc((size_t)NTOK * 2 * 4);
  float*          topk_w    = (float*)alloc((size_t)NTOK * 2 * 4);
  int*            tokrow    = (int*)alloc((size_t)NTOK * 2 * 4);
  int*            row2token = (int*)alloc((size_t)PAD_ROWS * 4);
  int*            counts    = (int*)alloc(64);
  int*            offv      = (int*)alloc(64);
  int*            cursor    = (int*)alloc(64);

  moe_init<<<(PAD_ROWS + 255) / 256, 256, 0, stream>>>(row2token, counts);
  moe_router<<<NTOK / 8, 256, 0, stream>>>(x, rw, xb, topk_idx, topk_w, counts);
  moe_wconv<<<dim3(H / 32, H / 32, E), dim3(32, 8), 0, stream>>>(w, wt);
  moe_scan<<<1, 32, 0, stream>>>(counts, offv, cursor);
  moe_scatter<<<(NTOK + 255) / 256, 256, 0, stream>>>(topk_idx, cursor,
                                                      row2token, tokrow);
  moe_gemm<<<ROW_TILES, 128, 0, stream>>>(xb, wt, offv, row2token, yp);
  moe_combine<<<NTOK, 128, 0, stream>>>(yp, tokrow, topk_w, out);
}
